// LSTM_35012573397358
// MI455X (gfx1250) — matrix operand order repaired and checked by reading, not compiled, not run on hardware
//
#include <hip/hip_runtime.h>
#include <hip/hip_bf16.h>

typedef float    v8f  __attribute__((ext_vector_type(8)));
typedef _Float16 v16h __attribute__((ext_vector_type(16)));
typedef _Float16 v8h  __attribute__((ext_vector_type(8)));

#define S_DIM 96
#define B_DIM 32
#define I_DIM 64
#define H_DIM 256

// ---------------------------------------------------------------------------
// Kernel 1: fx = x @ W_fx^T + b_fx   (S*B rows, H outputs each, K = I = 64)
// One block per (s,b) row; thread h computes one output element.
// ---------------------------------------------------------------------------
__global__ __launch_bounds__(256) void fx_proj_kernel(
    const float* __restrict__ x, const float* __restrict__ W,
    const float* __restrict__ b, float* __restrict__ out)
{
    __shared__ float xs[I_DIM];
    const int row = blockIdx.x;       // 0 .. S*B-1
    const int h   = threadIdx.x;      // 0 .. 255
    if (h < I_DIM) xs[h] = x[row * I_DIM + h];
    __syncthreads();
    float acc = b[h];
    const float* wr = W + h * I_DIM;
#pragma unroll
    for (int i = 0; i < I_DIM; ++i) acc = fmaf(xs[i], wr[i], acc);
    out[row * H_DIM + h] = acc;
}

// A-fragment loader per CDNA5 ISA 16-bit A 16x32 layout:
// lanes 0-15 (row m): K = kb+{0..7} in v0..v3, kb+{16..23} in v4..v7
// lanes 16-31 (row m): K = kb+{8..15}, kb+{24..31}
__device__ __forceinline__ v16h load_a_frag(const _Float16* __restrict__ row,
                                            int kb, int half8)
{
    const v8h lo = *(const v8h*)(row + kb + half8);
    const v8h hi = *(const v8h*)(row + kb + 16 + half8);
    return __builtin_shufflevector(lo, hi, 0, 1, 2, 3, 4, 5, 6, 7,
                                   8, 9, 10, 11, 12, 13, 14, 15);
}

// ---------------------------------------------------------------------------
// Kernel 2: one workgroup per chain s. 512 threads = 16 waves; wave w owns
// N-tile w (16 output columns) for both M-tiles (32 batch rows) and both
// recurrent GEMMs (g-gate and candidate). Weights resident in LDS as f16.
// ---------------------------------------------------------------------------
__global__ __launch_bounds__(512, 1) void lstm_chain_kernel(
    const float* __restrict__ x,
    const float* __restrict__ W_fh, const float* __restrict__ b_fh,
    const float* __restrict__ W_cx, const float* __restrict__ b_cx,
    const float* __restrict__ W_ch, const float* __restrict__ b_ch,
    const float* __restrict__ fx_ws, float* __restrict__ out)
{
    extern __shared__ char smem[];
    _Float16* Wf = (_Float16*)smem;                  // 256*256 f16 = 128 KB
    _Float16* Wc = Wf + H_DIM * H_DIM;               // 256*256 f16 = 128 KB
    float*    cx = (float*)(Wc + H_DIM * H_DIM);     // 32*256 f32  =  32 KB
    _Float16* hh = (_Float16*)(cx + B_DIM * H_DIM);  // 32*256 f16  =  16 KB

    const int s    = blockIdx.x;
    const int tid  = threadIdx.x;
    const int lane = tid & 31;
    const int wave = tid >> 5;          // 0..15 -> N tile index
    const int half = lane >> 4;         // 0/1 lane half
    const int nloc = lane & 15;
    const int ncol = wave * 16 + nloc;  // output column this lane owns

    // Stage recurrent weights into LDS (f16). B[h][n] = W[n][h] means lane n
    // reads ROW n of W, so no transpose needed: store W row-major as-is.
    for (int i = tid; i < H_DIM * H_DIM; i += 512) {
        Wf[i] = (_Float16)W_fh[i];
        Wc[i] = (_Float16)W_ch[i];
    }
    // cx[s] = x[s] @ W_cx^T + b_cx, computed once per chain into LDS. Also
    // zero-init the f16 recurrent state hh.
    for (int e = tid; e < B_DIM * H_DIM; e += 512) {
        const int bb = e >> 8, hcol = e & 255;
        float acc = b_cx[hcol];
        const float* xr = x + (s * B_DIM + bb) * I_DIM;
        const float* wr = W_cx + hcol * I_DIM;
#pragma unroll
        for (int i = 0; i < I_DIM; ++i) acc = fmaf(xr[i], wr[i], acc);
        cx[e] = acc;
        hh[e] = (_Float16)0.0f;
    }
    __syncthreads();

    const float bfv = b_fh[ncol];
    const float bcv = b_ch[ncol];

    // Cell state + last hidden value live in registers: this wave's WMMA C/D
    // tile positions are exactly the (m,n) elements it updates elementwise.
    float cc[2][8];
    float hv[2][8];
#pragma unroll
    for (int mt = 0; mt < 2; ++mt)
#pragma unroll
        for (int j = 0; j < 8; ++j) { cc[mt][j] = 0.0f; hv[mt][j] = 0.0f; }

    const int rowA  = lane & 15;   // M row within tile for A fragments
    const int half8 = half * 8;
    const int half16 = half * 16;

    for (int t = 0; t < S_DIM; ++t) {
        v8f accF[2] = {};
        v8f accC[2] = {};
#pragma unroll
        for (int kt = 0; kt < H_DIM / 32; ++kt) {
            const int kb = kt * 32;
            const v16h a0 = load_a_frag(hh + (0  + rowA) * H_DIM, kb, half8);
            const v16h a1 = load_a_frag(hh + (16 + rowA) * H_DIM, kb, half8);
            // B 32x16 layout matches A: lanes 0-15 hold K=kb+{0..7},kb+{16..23},
            // lanes 16-31 hold K=kb+{8..15},kb+{24..31}.
            const v16h bF = load_a_frag(Wf + ncol * H_DIM, kb, half8);
            const v16h bC = load_a_frag(Wc + ncol * H_DIM, kb, half8);
            accF[0] = __builtin_amdgcn_wmma_f32_16x16x32_f16(
                false, a0, false, bF, (short)0, accF[0], false, false);
            accF[1] = __builtin_amdgcn_wmma_f32_16x16x32_f16(
                false, a1, false, bF, (short)0, accF[1], false, false);
            accC[0] = __builtin_amdgcn_wmma_f32_16x16x32_f16(
                false, a0, false, bC, (short)0, accC[0], false, false);
            accC[1] = __builtin_amdgcn_wmma_f32_16x16x32_f16(
                false, a1, false, bC, (short)0, accC[1], false, false);
        }
        __syncthreads();   // all reads of hh (step t-1 state) complete

        const float* fxp  = fx_ws + (size_t)t * B_DIM * H_DIM;
        float*       outp = out + (((size_t)t * S_DIM + s) * B_DIM) * H_DIM;
#pragma unroll
        for (int mt = 0; mt < 2; ++mt) {
#pragma unroll
            for (int j = 0; j < 8; ++j) {
                const int m = mt * 16 + half8 + j;   // C/D: VGPR j, lane half
                const float zf = accF[mt][j] + fxp[m * H_DIM + ncol] + bfv;
                const float zc = accC[mt][j] + cx[m * H_DIM + ncol] + bcv;
                const float g    = 1.0f / (1.0f + __expf(-zf));
                const float cand = tanhf(zc);
                const float c    = g * (cc[mt][j] + cand);
                cc[mt][j] = c;
                const float hval = g * tanhf(c);
                hv[mt][j] = hval;
                hh[m * H_DIM + ncol]   = (_Float16)hval;  // next-step A matrix
                outp[m * H_DIM + ncol] = hval;            // h_seq[t][s]
            }
        }
        __syncthreads();   // new hh visible before next step's WMMA reads
    }

    // h_fin = h_seq[S-1], c_fin = final cell state.
    float* hfin = out + (size_t)S_DIM * S_DIM * B_DIM * H_DIM;
    float* cfin = hfin + (size_t)S_DIM * B_DIM * H_DIM;
#pragma unroll
    for (int mt = 0; mt < 2; ++mt)
#pragma unroll
        for (int j = 0; j < 8; ++j) {
            const int m = mt * 16 + half8 + j;
            const size_t o = ((size_t)s * B_DIM + m) * H_DIM + ncol;
            hfin[o] = hv[mt][j];
            cfin[o] = cc[mt][j];
        }
}

extern "C" void kernel_launch(void* const* d_in, const int* in_sizes, int n_in,
                              void* d_out, int out_size, void* d_ws, size_t ws_size,
                              hipStream_t stream) {
    const float* x    = (const float*)d_in[0];
    const float* W_fx = (const float*)d_in[1];
    const float* b_fx = (const float*)d_in[2];
    const float* W_fh = (const float*)d_in[3];
    const float* b_fh = (const float*)d_in[4];
    const float* W_cx = (const float*)d_in[5];
    const float* b_cx = (const float*)d_in[6];
    const float* W_ch = (const float*)d_in[7];
    const float* b_ch = (const float*)d_in[8];
    float* out   = (float*)d_out;
    float* fx_ws = (float*)d_ws;   // S*B*H f32 = 3 MB scratch

    // Input projection fx (loop-invariant, shared by all 96 chains).
    fx_proj_kernel<<<S_DIM * B_DIM, 256, 0, stream>>>(x, W_fx, b_fx, fx_ws);

    // One workgroup per independent recurrent chain; 304 KB LDS per WGP.
    const size_t lds_bytes = (size_t)2 * H_DIM * H_DIM * sizeof(_Float16)
                           + (size_t)B_DIM * H_DIM * sizeof(float)
                           + (size_t)B_DIM * H_DIM * sizeof(_Float16);
    lstm_chain_kernel<<<S_DIM, 512, lds_bytes, stream>>>(
        x, W_fh, b_fh, W_cx, b_cx, W_ch, b_ch, fx_ws, out);
}
